// SimpleBEVDecoder_20675972563510
// MI455X (gfx1250) — matrix operand order repaired and checked by reading, not compiled, not run on hardware
//
#include <hip/hip_runtime.h>
#include <hip/hip_bf16.h>
#include <cstdint>

// ---------------------------------------------------------------------------
// SimpleBEVDecoder for MI455X (gfx1250, wave32, WMMA).
//
// Pipeline:
//   0. weight prep: fp32 -> bf16 (+transpose PE weights to [Nout][Kin])
//   1. feat [6,128,32,88] -> channel-last fmap [6,32,88,128] (coalesced gathers)
//   2. sample: project 25600*4 ref points through 6 cameras, bilinear gather,
//      sum over cameras -> x0f [Q][512] fp32  (channel idx = p*128+c)
//   3. PE layer1 (K=3, scalar FMA) -> h1 [102400][256] bf16
//   4. PE layer2: WMMA bf16 GEMM + ReLU -> h2 bf16
//   5. PE layer3: WMMA GEMM, epilogue adds x0f, emits x0bf [Q][512] bf16
//   6. conv1 512x512 WMMA GEMM -> y1f; IN stats; IN+GELU -> y1bf
//   7. conv2 128x512 WMMA GEMM -> y2f; IN stats; IN+GELU -> y2bf
//   8. conv3 128x128 WMMA GEMM + bias, transposed store -> out [128][25600]
//
// GEMM: v_wmma_f32_16x16x32_bf16, 32-row block tile staged in LDS (async
// global->LDS B128 when the toolchain exposes the builtin), each wave owns
// one 16-col N-tile and two M-tiles so every weight fragment feeds 2 WMMAs.
// ---------------------------------------------------------------------------

typedef __attribute__((ext_vector_type(16))) __bf16 bf16x16;
typedef __attribute__((ext_vector_type(8)))  float  f32x8;
typedef unsigned short u16;
typedef unsigned int   u32;
typedef int v4i __attribute__((vector_size(16)));   // int4 as builtin expects

static constexpr int QTOT   = 25600;    // 160*160 BEV queries
static constexpr int QPROWS = 102400;   // Q * P
static constexpr int HB = 160, WB = 160;
static constexpr int HF = 32,  WF = 88;
static constexpr int NCAM = 6, CFEAT = 128;

#if defined(__AMDGCN__) && __has_builtin(__builtin_amdgcn_global_load_async_to_lds_b128)
#define HAVE_ASYNC_LDS 1
typedef __attribute__((address_space(1))) v4i* as1_v4i;
typedef __attribute__((address_space(3))) v4i* as3_v4i;
#else
#define HAVE_ASYNC_LDS 0
#endif

static __device__ __forceinline__ void wait_async0() {
#if defined(__AMDGCN__)
#if __has_builtin(__builtin_amdgcn_s_wait_asynccnt)
    __builtin_amdgcn_s_wait_asynccnt(0);
#else
    asm volatile("s_wait_asynccnt 0x0" ::: "memory");
#endif
#endif
}

static __device__ __forceinline__ u16 f2bf(float f) {
    u32 u = __float_as_uint(f);
    u32 r = u + 0x7FFFu + ((u >> 16) & 1u);   // round-to-nearest-even
    return (u16)(r >> 16);
}

// ---------------------------- prep kernels ---------------------------------

__global__ void k_cvt_bf16(const float* __restrict__ s, u16* __restrict__ d, int n) {
    int i = blockIdx.x * blockDim.x + threadIdx.x;
    if (i < n) d[i] = f2bf(s[i]);
}

// s: [K][N] fp32  ->  d: [N][K] bf16   (canonical W[Nout][Kin] layout)
__global__ void k_transpose_cvt(const float* __restrict__ s, u16* __restrict__ d,
                                int K, int N) {
    int i = blockIdx.x * blockDim.x + threadIdx.x;
    if (i >= K * N) return;
    int k = i % K, o = i / K;
    d[i] = f2bf(s[(size_t)k * N + o]);
}

// feat [6][128][32][88] -> fmap [6][32][88][128] (channel-last)
__global__ void k_transpose_feat(const float* __restrict__ feat, float* __restrict__ fmap) {
    int i = blockIdx.x * blockDim.x + threadIdx.x;
    const int total = NCAM * HF * WF * CFEAT;
    if (i >= total) return;
    int c = i & (CFEAT - 1);
    int t = i >> 7;            // n*HF*WF + y*WF + x
    int x = t % WF; t /= WF;
    int y = t % HF;
    int n = t / HF;
    fmap[i] = feat[(((size_t)n * CFEAT + c) * HF + y) * WF + x];
}

// ------------------------- projection + sampling ---------------------------

__global__ void k_sample(const float* __restrict__ fmap, const float* __restrict__ l2i,
                         const float* __restrict__ bev, float* __restrict__ x0f) {
    const int r = blockIdx.x;          // q*4 + p
    const int c = threadIdx.x;         // 0..127 (channel)
    const int q = r >> 2, p = r & 3;
    const int h = q / WB, w = q % WB;
    const float* bp = bev + (((size_t)p * HB + h) * WB + w) * 3;
    const float rx = bp[0], ry = bp[1], rz = bp[2];
    const float X = rx * 100.f - 50.f;
    const float Y = ry * 100.f - 50.f;
    const float Z = rz * 8.f - 4.f;
    float acc = 0.f;
    #pragma unroll
    for (int n = 0; n < NCAM; ++n) {
        const float* L = l2i + n * 16;
        float cx = L[0]*X + L[1]*Y + L[2]*Z  + L[3];
        float cy = L[4]*X + L[5]*Y + L[6]*Z  + L[7];
        float cz = L[8]*X + L[9]*Y + L[10]*Z + L[11];
        float z  = fmaxf(cz, 1e-5f);
        float u  = cx / (z * 704.f);
        float v  = cy / (z * 256.f);
        if (cz > 1e-5f && u > 0.f && u < 1.f && v > 0.f && v < 1.f) {
            float xf = u * (float)WF - 0.5f;
            float yf = v * (float)HF - 0.5f;
            float x0 = floorf(xf), y0 = floorf(yf);
            float dx = xf - x0,   dy = yf - y0;
            int xi = (int)x0, yi = (int)y0;
            float w00 = (1.f - dx) * (1.f - dy), w10 = dx * (1.f - dy);
            float w01 = (1.f - dx) * dy,         w11 = dx * dy;
            const float* base = fmap + (size_t)n * HF * WF * CFEAT;
            if (xi   >= 0 && xi   < WF && yi   >= 0 && yi   < HF)
                acc += base[((size_t)yi * WF + xi) * CFEAT + c] * w00;
            if (xi+1 >= 0 && xi+1 < WF && yi   >= 0 && yi   < HF)
                acc += base[((size_t)yi * WF + xi + 1) * CFEAT + c] * w10;
            if (xi   >= 0 && xi   < WF && yi+1 >= 0 && yi+1 < HF)
                acc += base[((size_t)(yi+1) * WF + xi) * CFEAT + c] * w01;
            if (xi+1 >= 0 && xi+1 < WF && yi+1 >= 0 && yi+1 < HF)
                acc += base[((size_t)(yi+1) * WF + xi + 1) * CFEAT + c] * w11;
        }
    }
    x0f[(size_t)q * 512 + p * 128 + c] = acc;
}

// ------------------------- PE layer1 (K=3, scalar) -------------------------

__global__ void k_pe_l1(const float* __restrict__ bev, const float* __restrict__ w1,
                        const float* __restrict__ b1, u16* __restrict__ h1) {
    const int r = blockIdx.x;          // row = q*4 + p  (matches [B,Q,P,*])
    const int j = threadIdx.x;         // 0..255
    const int q = r >> 2, p = r & 3;
    const int h = q / WB, w = q % WB;
    const float* bp = bev + (((size_t)p * HB + h) * WB + w) * 3;
    float s = bp[0] * w1[j] + bp[1] * w1[256 + j] + bp[2] * w1[512 + j] + b1[j];
    h1[(size_t)r * 256 + j] = f2bf(fmaxf(s, 0.f));
}

// ----------------------------- WMMA GEMM -----------------------------------
// Y[M][NOUT] = X[M][KIN](bf16) * W[NOUT][KIN]^T (bf16) + bias, f32 accumulate.
// 256 threads = 8 waves; block stages a 32-row M-tile in LDS; each wave owns
// one 16-col N-tile (grid.y * 8 + wave) and BOTH 16-row M-subtiles, so every
// B (weight) fragment is reused by 2 WMMAs.

enum { EPI_F32 = 0, EPI_RELU_BF16 = 1, EPI_ADD_X0 = 2, EPI_OUT_T = 3 };

template<int KIN, int NOUT, int EPI>
__global__ __launch_bounds__(256) void k_gemm(const u16* __restrict__ X,
                                              const u16* __restrict__ W,
                                              const float* __restrict__ bias,
                                              float* __restrict__ outf,
                                              u16* __restrict__ outbf,
                                              const float* __restrict__ addsrc) {
    __shared__ __align__(16) u16 Xs[32 * KIN];
    const int tid  = threadIdx.x;
    const int row0 = blockIdx.x * 32;
    constexpr int NV4 = (32 * KIN) / 8;   // 8 bf16 per 16-byte chunk

    // 32 consecutive rows of X are contiguous in memory -> bulk copy to LDS.
#if HAVE_ASYNC_LDS
    {
        as1_v4i gp = (as1_v4i)(v4i*)(void*)const_cast<u16*>(X + (size_t)row0 * KIN);
        as3_v4i lp = (as3_v4i)(v4i*)(void*)Xs;
        for (int i = tid; i < NV4; i += 256)
            __builtin_amdgcn_global_load_async_to_lds_b128(gp + i, lp + i, 0, 0);
        wait_async0();
    }
#else
    {
        const uint4* xg = reinterpret_cast<const uint4*>(X + (size_t)row0 * KIN);
        uint4* xs = reinterpret_cast<uint4*>(Xs);
        for (int i = tid; i < NV4; i += 256) xs[i] = xg[i];
    }
#endif
    __syncthreads();

    const int wave = tid >> 5;
    const int lane = tid & 31;
    const int g    = (lane >> 4) & 1;     // half-wave selector
    const int lm   = lane & 15;
    const int n0   = (blockIdx.y * 8 + wave) * 16;
    const int n    = n0 + lm;

    // A fragment: lane holds row M=lm; contiguous K order (matches B):
    //   VGPR r : K = k0 + 16g + 2r + h
    // -> vectorizes to two ds_load_b128 per fragment.
    const u32* xrow0 = reinterpret_cast<const u32*>(Xs + (size_t)lm * KIN);
    const u32* xrow1 = reinterpret_cast<const u32*>(Xs + (size_t)(lm + 16) * KIN);
    // B fragment: lane holds column N=n; element e -> K = k0 + e + 16g
    const u16* wrow = W + (size_t)n * KIN;

    f32x8 acc0, acc1;
    const float bv = bias ? bias[n] : 0.f;
    #pragma unroll
    for (int i = 0; i < 8; ++i) { acc0[i] = bv; acc1[i] = bv; }

    #pragma unroll 2
    for (int k0 = 0; k0 < KIN; k0 += 32) {
        union { bf16x16 v; u32 u[8]; } a0, a1, b;
        #pragma unroll
        for (int r = 0; r < 8; ++r) {
            int kp = k0 + 16*g + 2*r;
            a0.u[r] = xrow0[kp >> 1];          // pair (kp, kp+1), 32-bit aligned
            a1.u[r] = xrow1[kp >> 1];
        }
        const uint4* wp = reinterpret_cast<const uint4*>(wrow + k0 + 16*g);
        uint4 w0 = wp[0], w1 = wp[1];
        b.u[0] = w0.x; b.u[1] = w0.y; b.u[2] = w0.z; b.u[3] = w0.w;
        b.u[4] = w1.x; b.u[5] = w1.y; b.u[6] = w1.z; b.u[7] = w1.w;
        if (k0 + 32 < KIN) __builtin_prefetch(wrow + k0 + 48, 0, 3);
        acc0 = __builtin_amdgcn_wmma_f32_16x16x32_bf16(
                   false, a0.v, false, b.v, (short)0, acc0, false, false);
        acc1 = __builtin_amdgcn_wmma_f32_16x16x32_bf16(
                   false, a1.v, false, b.v, (short)0, acc1, false, false);
    }

    // C/D layout: VGPR v -> row M = v + 8g (lanes 0-15 / 16-31), col N = lm
    #pragma unroll
    for (int v = 0; v < 8; ++v) {
        const int ra = row0 + v + 8*g;        // M-subtile 0
        const int rb = ra + 16;               // M-subtile 1
        const float va = acc0[v];
        const float vb = acc1[v];
        if (EPI == EPI_F32) {
            outf[(size_t)ra * NOUT + n] = va;
            outf[(size_t)rb * NOUT + n] = vb;
        } else if (EPI == EPI_RELU_BF16) {
            outbf[(size_t)ra * NOUT + n] = f2bf(fmaxf(va, 0.f));
            outbf[(size_t)rb * NOUT + n] = f2bf(fmaxf(vb, 0.f));
        } else if (EPI == EPI_ADD_X0) {
            // row = q*4+p ; fuse PE output into sampled features, emit bf16
            size_t ia = (size_t)(ra >> 2) * 512 + (size_t)(ra & 3) * 128 + n;
            size_t ib = (size_t)(rb >> 2) * 512 + (size_t)(rb & 3) * 128 + n;
            outbf[ia] = f2bf(addsrc[ia] + va);
            outbf[ib] = f2bf(addsrc[ib] + vb);
        } else { // EPI_OUT_T : final output is [E][Q]
            outf[(size_t)n * QTOT + ra] = va;
            outf[(size_t)n * QTOT + rb] = vb;
        }
    }
}

// -------------------- instance norm (stats + apply+GELU) -------------------

template<int C>
__global__ void k_stats(const float* __restrict__ Y, float* __restrict__ mu,
                        float* __restrict__ rs) {
    __shared__ float ss[256], sq[256];
    const int c = blockIdx.x, tid = threadIdx.x;
    float s = 0.f, s2 = 0.f;
    for (int q = tid; q < QTOT; q += 256) {
        float v = Y[(size_t)q * C + c];
        s += v; s2 += v * v;
    }
    ss[tid] = s; sq[tid] = s2;
    __syncthreads();
    for (int w = 128; w > 0; w >>= 1) {
        if (tid < w) { ss[tid] += ss[tid + w]; sq[tid] += sq[tid + w]; }
        __syncthreads();
    }
    if (tid == 0) {
        float m   = ss[0] / (float)QTOT;
        float var = sq[0] / (float)QTOT - m * m;
        mu[c] = m;
        rs[c] = rsqrtf(var + 1e-5f);
    }
}

__global__ void k_inorm_gelu(const float* __restrict__ Y, const float* __restrict__ mu,
                             const float* __restrict__ rs, u16* __restrict__ out,
                             int C, int total) {
    int i = blockIdx.x * blockDim.x + threadIdx.x;
    if (i >= total) return;
    int c = i & (C - 1);                    // C is a power of two (512 / 128)
    float v = (Y[i] - mu[c]) * rs[c];
    float g = 0.5f * v * (1.f + erff(v * 0.70710678118654752f));  // exact GELU
    out[i] = f2bf(g);
}

// ------------------------------- launcher ----------------------------------

extern "C" void kernel_launch(void* const* d_in, const int* in_sizes, int n_in,
                              void* d_out, int out_size, void* d_ws, size_t ws_size,
                              hipStream_t stream) {
    (void)in_sizes; (void)n_in; (void)out_size; (void)ws_size;
    const float* feat  = (const float*)d_in[0];
    const float* l2i   = (const float*)d_in[1];
    const float* bev   = (const float*)d_in[2];
    const float* pe_w1 = (const float*)d_in[3];
    const float* pe_b1 = (const float*)d_in[4];
    const float* pe_w2 = (const float*)d_in[5];
    const float* pe_b2 = (const float*)d_in[6];
    const float* pe_w3 = (const float*)d_in[7];
    const float* pe_b3 = (const float*)d_in[8];
    const float* c1_w  = (const float*)d_in[9];
    const float* c1_b  = (const float*)d_in[10];
    const float* c2_w  = (const float*)d_in[11];
    const float* c2_b  = (const float*)d_in[12];
    const float* c3_w  = (const float*)d_in[13];
    const float* c3_b  = (const float*)d_in[14];
    float* out = (float*)d_out;

    char* ws = (char*)d_ws;
    size_t off = 0;
    auto take = [&](size_t bytes) -> char* {
        char* p = ws + off;
        off += (bytes + 255) & ~(size_t)255;
        return p;
    };

    const size_t FMAP_ELTS = (size_t)NCAM * HF * WF * CFEAT;   // 2,162,688
    float* fmap = (float*)take(FMAP_ELTS * 4);
    u16* w2t  = (u16*)take(256 * 256 * 2);     // pe_w2^T  [256][256]
    u16* w3t  = (u16*)take(128 * 256 * 2);     // pe_w3^T  [128][256]
    u16* c1bf = (u16*)take(512 * 512 * 2);
    u16* c2bf = (u16*)take(128 * 512 * 2);
    u16* c3bf = (u16*)take(128 * 128 * 2);
    float* x0f = (float*)take((size_t)QTOT * 512 * 4);    // sampled feats; reused as y1f
    u16* h1    = (u16*)take((size_t)QPROWS * 256 * 2);    // reused as y1bf + y2bf
    u16* h2    = (u16*)take((size_t)QPROWS * 256 * 2);    // reused as y2f
    u16* x0bf  = (u16*)take((size_t)QTOT * 512 * 2);
    float* mu1 = (float*)take(512 * 4);
    float* rs1 = (float*)take(512 * 4);
    float* mu2 = (float*)take(128 * 4);
    float* rs2 = (float*)take(128 * 4);

    // buffer reuse (lifetimes are disjoint in launch order below)
    float* y1f  = x0f;                                        // 52.4 MB
    u16*   y1bf = h1;                                         // 26.2 MB
    float* y2f  = (float*)h2;                                 // 13.1 MB
    u16*   y2bf = (u16*)((char*)h1 + (size_t)QTOT * 512 * 2); // 6.6 MB

    // 0) weight prep
    k_transpose_cvt<<<(256*256 + 255)/256, 256, 0, stream>>>(pe_w2, w2t, 256, 256);
    k_transpose_cvt<<<(256*128 + 255)/256, 256, 0, stream>>>(pe_w3, w3t, 256, 128);
    k_cvt_bf16<<<(512*512 + 255)/256, 256, 0, stream>>>(c1_w, c1bf, 512*512);
    k_cvt_bf16<<<(128*512 + 255)/256, 256, 0, stream>>>(c2_w, c2bf, 128*512);
    k_cvt_bf16<<<(128*128 + 255)/256, 256, 0, stream>>>(c3_w, c3bf, 128*128);

    // 1) feat -> channel-last
    k_transpose_feat<<<(int)((FMAP_ELTS + 255)/256), 256, 0, stream>>>(feat, fmap);

    // 2) projection + bilinear sampling -> x0f [Q][512]
    k_sample<<<QPROWS, 128, 0, stream>>>(fmap, l2i, bev, x0f);

    // 3) PE layer1 -> h1 [102400][256] bf16
    k_pe_l1<<<QPROWS, 256, 0, stream>>>(bev, pe_w1, pe_b1, h1);

    // 4) PE layer2: h2 = relu(h1 @ w2 + b2)
    k_gemm<256, 256, EPI_RELU_BF16><<<dim3(QPROWS/32, 2), 256, 0, stream>>>(
        h1, w2t, pe_b2, nullptr, h2, nullptr);

    // 5) PE layer3 fused with sampled features: x0bf = bf16(x0f + h2 @ w3 + b3)
    k_gemm<256, 128, EPI_ADD_X0><<<dim3(QPROWS/32, 1), 256, 0, stream>>>(
        h2, w3t, pe_b3, nullptr, x0bf, x0f);

    // 6) conv1 (512x512) -> y1f ; instance-norm ; GELU -> y1bf
    k_gemm<512, 512, EPI_F32><<<dim3(QTOT/32, 4), 256, 0, stream>>>(
        x0bf, c1bf, c1_b, y1f, nullptr, nullptr);
    k_stats<512><<<512, 256, 0, stream>>>(y1f, mu1, rs1);
    k_inorm_gelu<<<(QTOT*512 + 255)/256, 256, 0, stream>>>(y1f, mu1, rs1, y1bf,
                                                           512, QTOT*512);

    // 7) conv2 (128x512) -> y2f ; instance-norm ; GELU -> y2bf
    k_gemm<512, 128, EPI_F32><<<dim3(QTOT/32, 1), 256, 0, stream>>>(
        y1bf, c2bf, c2_b, y2f, nullptr, nullptr);
    k_stats<128><<<128, 256, 0, stream>>>(y2f, mu2, rs2);
    k_inorm_gelu<<<(QTOT*128 + 255)/256, 256, 0, stream>>>(y2f, mu2, rs2, y2bf,
                                                           128, QTOT*128);

    // 8) conv3 (128x128) + bias, transposed store -> out [128][25600]
    k_gemm<128, 128, EPI_OUT_T><<<dim3(QTOT/32, 1), 256, 0, stream>>>(
        y2bf, c3bf, c3_b, out, nullptr, nullptr);
}
